// LlamaAttentionBlock_59931973648867
// MI455X (gfx1250) — matrix operand order repaired and checked by reading, not compiled, not run on hardware
//
#include <hip/hip_runtime.h>

// Problem constants (B,T,C,H from the reference)
#define BB 4
#define TT 2048
#define CC 1024
#define HH 16
#define DD 64
#define BT (BB * TT)          // 8192 rows
#define N3C (3 * CC)          // 3072

typedef __attribute__((ext_vector_type(16))) __bf16 v16bf;
typedef __attribute__((ext_vector_type(8)))  float  v8f;
typedef __attribute__((ext_vector_type(4)))  unsigned int u32x4;
typedef __attribute__((ext_vector_type(8)))  int i32x8;
typedef __attribute__((ext_vector_type(4)))  int i32x4;

#if defined(__has_builtin)
#if __has_builtin(__builtin_amdgcn_tensor_load_to_lds) && \
    __has_builtin(__builtin_amdgcn_s_wait_tensorcnt)
#define USE_TDM 1
#endif
#endif

union FragCast { uint4 q[2]; v16bf b; };

__device__ __forceinline__ v16bf make_frag(uint4 a, uint4 c) {
  FragCast f; f.q[0] = a; f.q[1] = c; return f.b;
}

__device__ __forceinline__ unsigned short f32_bf16(float x) {
  unsigned int u = __builtin_bit_cast(unsigned int, x);
  u += 0x7FFFu + ((u >> 16) & 1u);          // round-to-nearest-even
  return (unsigned short)(u >> 16);
}

__device__ __forceinline__ v8f zero_v8f() {
  v8f z;
#pragma unroll
  for (int i = 0; i < 8; ++i) z[i] = 0.0f;
  return z;
}

// reductions across the 16 lanes that share one C-fragment row (wave32)
__device__ __forceinline__ float redmax16(float v) {
  v = fmaxf(v, __shfl_xor(v, 1, 32));
  v = fmaxf(v, __shfl_xor(v, 2, 32));
  v = fmaxf(v, __shfl_xor(v, 4, 32));
  v = fmaxf(v, __shfl_xor(v, 8, 32));
  return v;
}
__device__ __forceinline__ float redsum16(float v) {
  v += __shfl_xor(v, 1, 32);
  v += __shfl_xor(v, 2, 32);
  v += __shfl_xor(v, 4, 32);
  v += __shfl_xor(v, 8, 32);
  return v;
}

// ---------------- precision conversion kernels ----------------

__global__ void __launch_bounds__(256) cvt_bf16_kernel(
    const float* __restrict__ in, unsigned short* __restrict__ out, int n) {
  for (int i = blockIdx.x * blockDim.x + threadIdx.x; i < n;
       i += gridDim.x * blockDim.x)
    out[i] = f32_bf16(in[i]);
}

// w: [K][N] row-major f32  ->  wT: [N][K] bf16 (so B-fragments are K-contiguous)
__global__ void __launch_bounds__(256) transpose_cvt_kernel(
    const float* __restrict__ w, unsigned short* __restrict__ wT, int K, int N) {
  int total = K * N;
  for (int i = blockIdx.x * blockDim.x + threadIdx.x; i < total;
       i += gridDim.x * blockDim.x) {
    int n = i / K;
    int k = i - n * K;
    wT[i] = f32_bf16(w[(size_t)k * N + n]);
  }
}

// ---------------- QKV projection GEMM ----------------
// One wave computes a 16(M) x 64(N) strip. Explicit two-slot ping-pong keeps
// each fragment in fixed registers (no rotation copies) while the next
// k-step's loads overlap the WMMA pipe.
__global__ void __launch_bounds__(256) gemm_qkv_kernel(
    const unsigned short* __restrict__ hb,    // [BT][CC] bf16
    const unsigned short* __restrict__ waT,   // [3C][CC] bf16 (transposed w_attn)
    unsigned short* __restrict__ qb,
    unsigned short* __restrict__ kb,
    unsigned short* __restrict__ vb) {
  const int lane = threadIdx.x & 31;
  const int wave = threadIdx.x >> 5;
  const int widx = blockIdx.x * 8 + wave;     // 24576 waves total
  const int nt = widx % 48;                   // 64-wide N group
  const int mt = widx / 48;                   // 16-row M tile
  const int g  = lane >> 4;
  const int lr = lane & 15;

  const unsigned short* arow = hb + (size_t)(mt * 16 + lr) * CC + g * 8;
  const unsigned short* brow[4];
#pragma unroll
  for (int nn = 0; nn < 4; ++nn)
    brow[nn] = waT + (size_t)(nt * 64 + nn * 16 + lr) * CC + g * 8;

  v8f acc[4];
#pragma unroll
  for (int i = 0; i < 4; ++i) acc[i] = zero_v8f();

  uint4 a0[2], a1[2], b0[2][4], b1[2][4];
  auto loadslot = [&](int s, int k) {
    a0[s] = *(const uint4*)(arow + k);
    a1[s] = *(const uint4*)(arow + k + 16);
#pragma unroll
    for (int nn = 0; nn < 4; ++nn) {
      b0[s][nn] = *(const uint4*)(brow[nn] + k);
      b1[s][nn] = *(const uint4*)(brow[nn] + k + 16);
    }
  };
  auto compute = [&](int s) {
    v16bf af = make_frag(a0[s], a1[s]);
#pragma unroll
    for (int nn = 0; nn < 4; ++nn) {
      v16bf bf = make_frag(b0[s][nn], b1[s][nn]);
      acc[nn] = __builtin_amdgcn_wmma_f32_16x16x32_bf16(
          false, af, false, bf, (short)0, acc[nn], false, false);
    }
  };

  loadslot(0, 0);
  for (int kk = 0; kk < CC; kk += 64) {       // CC is a multiple of 64
    loadslot(1, kk + 32);
    compute(0);
    if (kk + 64 < CC) loadslot(0, kk + 64);
    compute(1);
  }

#pragma unroll
  for (int nn = 0; nn < 4; ++nn) {
    int nc  = nt * 64 + nn * 16 + lr;   // column in [0,3C)
    int sel = nc >> 10;                 // 0:q 1:k 2:v
    int c   = nc & (CC - 1);
    int hh  = c >> 6;                   // head (constant within 16-col subtile)
    int d   = c & 63;
    unsigned short* dst = (sel == 0) ? qb : (sel == 1) ? kb : vb;
#pragma unroll
    for (int r = 0; r < 8; ++r) {
      int grow = mt * 16 + r + g * 8;   // C-layout: row = r + g*8
      int bidx = grow >> 11;
      int t    = grow & (TT - 1);
      dst[((size_t)(bidx * HH + hh) * TT + t) * DD + d] = f32_bf16(acc[nn][r]);
    }
  }
}

// ---------------- Flash attention ----------------
// Block = 8 waves = 128 q rows of one (b,h). Keys processed in chunks of 32.
// K-chunk staged into LDS via the Tensor Data Mover (tensor_load_to_lds with
// D# pad fields reproducing the 72-element padded row stride); V^T staged by
// the block; P round-trips through per-wave LDS (C-layout -> A-layout).
__global__ void __launch_bounds__(256) flash_attn_kernel(
    const unsigned short* __restrict__ qb,   // [B*H][T][D] bf16
    const unsigned short* __restrict__ kb,
    const unsigned short* __restrict__ vb,
    const float* __restrict__ mask,          // [B][T] additive
    unsigned short* __restrict__ yb) {       // [B*T][C] bf16
  // padded strides keep 16B alignment and spread LDS banks
  __shared__ unsigned short Kt[32 * 72];     // [key 0..31][d 0..63] stride 72
  __shared__ unsigned short Vt[64 * 40];     // [d 0..63][key 0..31] stride 40
  __shared__ unsigned short Pb[8 * 16 * 40]; // per-wave [row 0..15][key 0..31]

  const int bh   = blockIdx.x;
  const int b    = bh >> 4;
  const int hh   = bh & 15;
  const int lane = threadIdx.x & 31;
  const int wave = threadIdx.x >> 5;
  const int g    = lane >> 4;
  const int lr   = lane & 15;
  const int q0   = blockIdx.y * 128 + wave * 16;

  // Q fragments stay in registers (16x64 = two 16x32 A-fragments)
  const unsigned short* qrow = qb + ((size_t)bh * TT + q0 + lr) * DD;
  v16bf qa[2];
#pragma unroll
  for (int kc = 0; kc < 2; ++kc)
    qa[kc] = make_frag(*(const uint4*)(qrow + kc * 32 + g * 8),
                       *(const uint4*)(qrow + kc * 32 + 16 + g * 8));

  v8f acc[4];
#pragma unroll
  for (int i = 0; i < 4; ++i) acc[i] = zero_v8f();
  float mrow[8], lrow[8];
#pragma unroll
  for (int r = 0; r < 8; ++r) { mrow[r] = -3.0e38f; lrow[r] = 0.0f; }

  unsigned short* Pw = Pb + wave * 16 * 40;
  const float* maskrow = mask + b * TT;

#ifdef USE_TDM
  // LDS byte offset of Kt: generic pointer to LDS has offset in ADDR[31:0]
  const unsigned kt_lds = (unsigned)(size_t)(&Kt[0]);
#endif

  for (int n0 = 0; n0 < TT; n0 += 32) {
    __syncthreads();   // previous chunk's readers are done

#ifdef USE_TDM
    if (wave == 0) {
      // --- Tensor DMA: K tile [32 keys][64 d] bf16 -> Kt with padded rows ---
      unsigned long long ga =
          (unsigned long long)(size_t)(kb + ((size_t)bh * TT + n0) * DD);
      u32x4 g0;
      g0[0] = 1u;                                    // count=1 valid descriptor
      g0[1] = kt_lds;                                // lds_addr (bytes)
      g0[2] = (unsigned)(ga & 0xFFFFFFFFu);          // global_addr[31:0]
      g0[3] = (unsigned)((ga >> 32) & 0x01FFFFFFu)   // global_addr[56:32]
              | (2u << 30);                          // type=2 ("image")
      i32x8 g1;
      g1[0] = (1 << 16)      // data_size = 1 -> 2 bytes
            | (1 << 20)      // pad_enable
            | (4 << 22)      // pad_interval: code 4 -> every 32 DWORDs (=64 elems)
            | (3 << 25);     // pad_amount:  code 3 -> 4 DWORDs (=8 elems)
      g1[1] = (DD & 0xFFFF) << 16;            // tensor_dim0 = 64 (low bits)
      g1[2] = ((TT - n0) & 0xFFFF) << 16;     // tensor_dim1 (rows from tile start)
      g1[3] = (DD & 0xFFFF) << 16;            // tile_dim0 = 64
      g1[4] = 32;                             // tile_dim1 = 32 keys
      g1[5] = DD;                             // tensor_dim0_stride = 64 elems
      g1[6] = 0;
      g1[7] = 0;
      i32x4 gz4; gz4[0] = 0; gz4[1] = 0; gz4[2] = 0; gz4[3] = 0;
      i32x8 gz8;
#pragma unroll
      for (int i = 0; i < 8; ++i) gz8[i] = 0;
      // 6-arg toolchain variant: (g0, g1, g2, g3, g4, cpol)
      __builtin_amdgcn_tensor_load_to_lds(g0, g1, gz4, gz4, gz8, 0);
      __builtin_amdgcn_s_wait_tensorcnt(0);
    }
#endif
    {
      // cooperative staging: 256 threads x 8 bf16
      int row = threadIdx.x >> 3;          // key index 0..31
      int col = (threadIdx.x & 7) * 8;     // d offset 0..56
      size_t src = ((size_t)bh * TT + n0 + row) * DD + col;
#ifndef USE_TDM
      *(uint4*)(&Kt[row * 72 + col]) = *(const uint4*)(kb + src);
#endif
      union { uint4 q; unsigned short s[8]; } u;
      u.q = *(const uint4*)(vb + src);
#pragma unroll
      for (int j = 0; j < 8; ++j) Vt[(col + j) * 40 + row] = u.s[j];
    }
    __syncthreads();

    // scores: two 16x16 tiles (keys n0..n0+15, n0+16..n0+31), K-dim = D = 64
    v8f s[2];
#pragma unroll
    for (int j = 0; j < 2; ++j) {
      s[j] = zero_v8f();
#pragma unroll
      for (int kc = 0; kc < 2; ++kc) {
        const unsigned short* kp = &Kt[(j * 16 + lr) * 72 + kc * 32 + g * 8];
        v16bf bf = make_frag(*(const uint4*)kp, *(const uint4*)(kp + 16));
        s[j] = __builtin_amdgcn_wmma_f32_16x16x32_bf16(
            false, qa[kc], false, bf, (short)0, s[j], false, false);
      }
    }

    float mk0 = maskrow[n0 + lr];
    float mk1 = maskrow[n0 + 16 + lr];

    // online softmax update per row (row r+g*8 lives across 16 lanes)
#pragma unroll
    for (int r = 0; r < 8; ++r) {
      float s0 = s[0][r] * 0.125f + mk0;   // 1/sqrt(64)
      float s1 = s[1][r] * 0.125f + mk1;
      float mx = redmax16(fmaxf(s0, s1));
      float mn = fmaxf(mrow[r], mx);
      float sc = __expf(mrow[r] - mn);
      float p0 = __expf(s0 - mn);
      float p1 = __expf(s1 - mn);
      float rs = redsum16(p0 + p1);
      lrow[r] = lrow[r] * sc + rs;
      mrow[r] = mn;
#pragma unroll
      for (int dd = 0; dd < 4; ++dd) acc[dd][r] *= sc;
      int prow = (r + g * 8) * 40;
      Pw[prow + lr]      = f32_bf16(p0);
      Pw[prow + 16 + lr] = f32_bf16(p1);
    }

    // reload P as a 16x32 A-fragment (same-wave LDS ops are in-order)
    const unsigned short* pr = &Pw[lr * 40];
    v16bf pa = make_frag(*(const uint4*)(pr + g * 8),
                         *(const uint4*)(pr + 16 + g * 8));
#pragma unroll
    for (int dd = 0; dd < 4; ++dd) {
      const unsigned short* vp = &Vt[(dd * 16 + lr) * 40 + g * 8];
      v16bf bf = make_frag(*(const uint4*)vp, *(const uint4*)(vp + 16));
      acc[dd] = __builtin_amdgcn_wmma_f32_16x16x32_bf16(
          false, pa, false, bf, (short)0, acc[dd], false, false);
    }
  }

  // finalize: y = acc / l, written straight into [B*T][C] bf16 layout
#pragma unroll
  for (int r = 0; r < 8; ++r) {
    float inv = 1.0f / lrow[r];
    int trow = q0 + r + g * 8;
    unsigned short* dst = yb + ((size_t)(b * TT + trow)) * CC + hh * DD;
#pragma unroll
    for (int dd = 0; dd < 4; ++dd) dst[dd * 16 + lr] = f32_bf16(acc[dd][r] * inv);
  }
}

// ---------------- output projection GEMM (f32 out) ----------------
__global__ void __launch_bounds__(256) gemm_proj_kernel(
    const unsigned short* __restrict__ yb,    // [BT][CC] bf16
    const unsigned short* __restrict__ wpT,   // [CC][CC] bf16 (transposed)
    float* __restrict__ out) {                // [BT][CC] f32
  const int lane = threadIdx.x & 31;
  const int wave = threadIdx.x >> 5;
  const int widx = blockIdx.x * 8 + wave;     // 8192 waves
  const int nt = widx % 16;
  const int mt = widx / 16;
  const int g  = lane >> 4;
  const int lr = lane & 15;

  const unsigned short* arow = yb + (size_t)(mt * 16 + lr) * CC + g * 8;
  const unsigned short* brow[4];
#pragma unroll
  for (int nn = 0; nn < 4; ++nn)
    brow[nn] = wpT + (size_t)(nt * 64 + nn * 16 + lr) * CC + g * 8;

  v8f acc[4];
#pragma unroll
  for (int i = 0; i < 4; ++i) acc[i] = zero_v8f();

  uint4 a0[2], a1[2], b0[2][4], b1[2][4];
  auto loadslot = [&](int s, int k) {
    a0[s] = *(const uint4*)(arow + k);
    a1[s] = *(const uint4*)(arow + k + 16);
#pragma unroll
    for (int nn = 0; nn < 4; ++nn) {
      b0[s][nn] = *(const uint4*)(brow[nn] + k);
      b1[s][nn] = *(const uint4*)(brow[nn] + k + 16);
    }
  };
  auto compute = [&](int s) {
    v16bf af = make_frag(a0[s], a1[s]);
#pragma unroll
    for (int nn = 0; nn < 4; ++nn) {
      v16bf bf = make_frag(b0[s][nn], b1[s][nn]);
      acc[nn] = __builtin_amdgcn_wmma_f32_16x16x32_bf16(
          false, af, false, bf, (short)0, acc[nn], false, false);
    }
  };

  loadslot(0, 0);
  for (int kk = 0; kk < CC; kk += 64) {
    loadslot(1, kk + 32);
    compute(0);
    if (kk + 64 < CC) loadslot(0, kk + 64);
    compute(1);
  }

#pragma unroll
  for (int nn = 0; nn < 4; ++nn) {
    int nc = nt * 64 + nn * 16 + lr;
#pragma unroll
    for (int r = 0; r < 8; ++r) {
      int grow = mt * 16 + r + g * 8;
      out[(size_t)grow * CC + nc] = acc[nn][r];
    }
  }
}

// ---------------- host-side launch ----------------
extern "C" void kernel_launch(void* const* d_in, const int* in_sizes, int n_in,
                              void* d_out, int out_size, void* d_ws, size_t ws_size,
                              hipStream_t stream) {
  (void)in_sizes; (void)n_in; (void)out_size; (void)ws_size;
  const float* h      = (const float*)d_in[0];
  const float* mask   = (const float*)d_in[1];
  const float* w_attn = (const float*)d_in[2];
  const float* w_proj = (const float*)d_in[3];
  float* out = (float*)d_out;

  unsigned char* ws = (unsigned char*)d_ws;
  size_t off = 0;
  auto wsalloc = [&](size_t bytes) -> void* {
    void* p = ws + off;
    off += (bytes + 255) & ~(size_t)255;
    return p;
  };
  unsigned short* hb   = (unsigned short*)wsalloc((size_t)BT * CC * 2);   // 16 MB
  unsigned short* waT  = (unsigned short*)wsalloc((size_t)N3C * CC * 2);  //  6 MB
  unsigned short* wpT  = (unsigned short*)wsalloc((size_t)CC * CC * 2);   //  2 MB
  unsigned short* qbuf = (unsigned short*)wsalloc((size_t)BT * CC * 2);   // 16 MB
  unsigned short* kbuf = (unsigned short*)wsalloc((size_t)BT * CC * 2);   // 16 MB
  unsigned short* vbuf = (unsigned short*)wsalloc((size_t)BT * CC * 2);   // 16 MB
  unsigned short* ybuf = (unsigned short*)wsalloc((size_t)BT * CC * 2);   // 16 MB

  cvt_bf16_kernel<<<2048, 256, 0, stream>>>(h, hb, BT * CC);
  transpose_cvt_kernel<<<2048, 256, 0, stream>>>(w_attn, waT, CC, N3C);
  transpose_cvt_kernel<<<1024, 256, 0, stream>>>(w_proj, wpT, CC, CC);

  gemm_qkv_kernel<<<3072, 256, 0, stream>>>(hb, waT, qbuf, kbuf, vbuf);
  flash_attn_kernel<<<dim3(BB * HH, TT / 128), 256, 0, stream>>>(
      qbuf, kbuf, vbuf, mask, ybuf);
  gemm_proj_kernel<<<1024, 256, 0, stream>>>(ybuf, wpT, out);
}
